// MultiHeadLatentAttention_63350767616600
// MI455X (gfx1250) — matrix operand order repaired and checked by reading, not compiled, not run on hardware
//
#include <hip/hip_runtime.h>
#include <hip/hip_bf16.h>
#include <math.h>

// ---------------- problem constants (match reference) ----------------
constexpr int kB      = 2;
constexpr int kS      = 2048;
constexpr int kDim    = 2048;
constexpr int kNH     = 16;
constexpr int kQRank  = 1536;
constexpr int kKVRank = 512;
constexpr int kDNope  = 128;
constexpr int kDRope  = 64;
constexpr int kDV     = 128;
constexpr int kDHead  = kDNope + kDRope;   // 192
constexpr int kRows   = kB * kS;           // 4096 token rows

// Async global->LDS path (CDNA5 ASYNCcnt); guarded so we fall back to the
// proven synchronous staging if this toolchain lacks the builtins.
#if defined(__gfx1250__)
#if __has_builtin(__builtin_amdgcn_global_load_async_to_lds_b128) && \
    __has_builtin(__builtin_amdgcn_s_wait_asynccnt)
#define USE_ASYNC_LDS 1
#endif
#endif

typedef __attribute__((ext_vector_type(16))) _Float16 v16h;
typedef __attribute__((ext_vector_type(8)))  _Float16 v8h;
typedef __attribute__((ext_vector_type(8)))  float    v8f;

#ifdef USE_ASYNC_LDS
// Builtin signature (from hipcc diagnostic): first param is
// 'int __vector(4) __device__ *' -> 128-bit int vector in AS1; LDS side AS3.
typedef int vsi4 __attribute__((vector_size(16)));
typedef __attribute__((address_space(1))) vsi4 as1_vsi4;
typedef __attribute__((address_space(3))) vsi4 as3_vsi4;
__device__ __forceinline__ void async_copy_b128(const void* g, void* l) {
  __builtin_amdgcn_global_load_async_to_lds_b128(
      (as1_vsi4*)(__attribute__((address_space(1))) void*)(void*)g,
      (as3_vsi4*)(__attribute__((address_space(3))) void*)l,
      0, 0);
}
#endif

__device__ __forceinline__ v8f zero8f() {
  v8f r;
#pragma unroll
  for (int i = 0; i < 8; ++i) r[i] = 0.0f;
  return r;
}
__device__ __forceinline__ v8h zero8h() {
  v8h r;
#pragma unroll
  for (int i = 0; i < 8; ++i) r[i] = (_Float16)0.0f;
  return r;
}
__device__ __forceinline__ v16h make_frag(v8h lo, v8h hi) {
  union { v16h v; v8h h[2]; } u;
  u.h[0] = lo; u.h[1] = hi;
  return u.v;
}

// ---------------- fp32 -> fp16 convert ----------------
__global__ void f32_to_f16_kernel(const float* __restrict__ src,
                                  _Float16* __restrict__ dst, size_t n) {
  size_t i = (size_t)blockIdx.x * blockDim.x + threadIdx.x;
  size_t stride = (size_t)gridDim.x * blockDim.x;
  for (; i < n; i += stride) dst[i] = (_Float16)src[i];
}

// ---------------- NT GEMM: C[M,N] = A[M,K] * B[N,K]^T ----------------
// A, B fp16 row-major over K. 128x128 tile, 8 waves, wave tile 32x64,
// K step 32 via v_wmma_f32_16x16x32_f16. Double-buffered LDS; tiles are
// fetched with GLOBAL_LOAD_ASYNC_TO_LDS_B128 (ASYNCcnt) when available so
// the async engine fills buffer k+1 while the WMMA pipes chew on buffer k.
template <bool OUT_F16>
__global__ __launch_bounds__(256)
void gemm_nt_wmma_kernel(const _Float16* __restrict__ A,
                         const _Float16* __restrict__ B,
                         void* __restrict__ Cout, int M, int N, int K) {
  __shared__ _Float16 As[2][128][32];
  __shared__ _Float16 Bs[2][128][32];

  const int tid  = threadIdx.x;
  const int lane = tid & 31;
  const int wave = tid >> 5;      // 0..7
  const int wm   = wave >> 1;     // 0..3  (32 rows each)
  const int wn   = wave & 1;      // 0..1  (64 cols each)
  const int m0   = blockIdx.y * 128;
  const int n0   = blockIdx.x * 128;

  v8f acc[2][4];
#pragma unroll
  for (int i = 0; i < 2; ++i)
#pragma unroll
    for (int j = 0; j < 4; ++j) acc[i][j] = zero8f();

  const int lrow = tid >> 2;         // 0..63
  const int lcol = (tid & 3) * 8;    // 0,8,16,24
  const int ml   = lane & 15;
  const int hi   = lane >> 4;

  // stage one 128x32 A tile and one 128x32 B tile into LDS buffer `buf`
  auto stage_tile = [&](int k0, int buf) {
#pragma unroll
    for (int h = 0; h < 2; ++h) {
      const int r  = lrow + h * 64;
      const int gm = m0 + r;
      const int gn = n0 + r;
#ifdef USE_ASYNC_LDS
      if (gm < M)
        async_copy_b128(A + (size_t)gm * K + k0 + lcol, &As[buf][r][lcol]);
      else
        *(v8h*)&As[buf][r][lcol] = zero8h();
      if (gn < N)
        async_copy_b128(B + (size_t)gn * K + k0 + lcol, &Bs[buf][r][lcol]);
      else
        *(v8h*)&Bs[buf][r][lcol] = zero8h();
#else
      v8h va = zero8h();
      if (gm < M) va = *(const v8h*)(A + (size_t)gm * K + k0 + lcol);
      *(v8h*)&As[buf][r][lcol] = va;
      v8h vb = zero8h();
      if (gn < N) vb = *(const v8h*)(B + (size_t)gn * K + k0 + lcol);
      *(v8h*)&Bs[buf][r][lcol] = vb;
      if (k0 + 32 < K) {
        if (gm < M)
          __builtin_prefetch(A + (size_t)gm * K + k0 + 32 + lcol, 0, 1);
        if (gn < N)
          __builtin_prefetch(B + (size_t)gn * K + k0 + 32 + lcol, 0, 1);
      }
#endif
    }
  };

  stage_tile(0, 0);
#ifdef USE_ASYNC_LDS
  __builtin_amdgcn_s_wait_asynccnt(0);
#endif
  __syncthreads();

  int buf = 0;
  for (int k0 = 0; k0 < K; k0 += 32) {
    // kick off the next tile into the other buffer (overlaps with WMMAs)
    if (k0 + 32 < K) stage_tile(k0 + 32, buf ^ 1);

    // A fragments: lane (m = ml), K chunks kh*8 and 16+kh*8  (ISA 7.12.2)
    const int kh8 = hi * 8;
    v16h afrag[2];
#pragma unroll
    for (int i = 0; i < 2; ++i) {
      const _Float16* p = &As[buf][wm * 32 + i * 16 + ml][0];
      afrag[i] = make_frag(*(const v8h*)(p + kh8), *(const v8h*)(p + 16 + kh8));
    }
    // B fragments: lane (n = ml), K chunks kh*8 and 16+kh*8
    const int ks16 = hi * 16;
    v16h bfrag[4];
#pragma unroll
    for (int j = 0; j < 4; ++j) {
      const _Float16* p = &Bs[buf][wn * 64 + j * 16 + ml][0];
      bfrag[j] = make_frag(*(const v8h*)(p + kh8), *(const v8h*)(p + 16 + kh8));
    }
#pragma unroll
    for (int i = 0; i < 2; ++i)
#pragma unroll
      for (int j = 0; j < 4; ++j)
        acc[i][j] = __builtin_amdgcn_wmma_f32_16x16x32_f16(
            false, afrag[i], false, bfrag[j], (short)0, acc[i][j], false, false);

#ifdef USE_ASYNC_LDS
    __builtin_amdgcn_s_wait_asynccnt(0);
#endif
    __syncthreads();
    buf ^= 1;
  }

  // store: element s of acc is row (s + 8*hi), col (ml) of the 16x16 tile
#pragma unroll
  for (int i = 0; i < 2; ++i)
#pragma unroll
    for (int j = 0; j < 4; ++j)
#pragma unroll
      for (int s = 0; s < 8; ++s) {
        const int gm = m0 + wm * 32 + i * 16 + s + 8 * hi;
        const int gn = n0 + wn * 64 + j * 16 + ml;
        if (gm < M && gn < N) {
          if (OUT_F16)
            ((_Float16*)Cout)[(size_t)gm * N + gn] = (_Float16)acc[i][j][s];
          else
            ((float*)Cout)[(size_t)gm * N + gn] = acc[i][j][s];
        }
      }
}

// ---------------- RMS norm over rows (fp32 in, fp16 out) ----------------
__global__ __launch_bounds__(256)
void rmsnorm_kernel(const float* __restrict__ X, const float* __restrict__ W,
                    _Float16* __restrict__ Y, int R) {
  __shared__ float red[256];
  const int row = blockIdx.x;
  const int tid = threadIdx.x;
  const float* xr = X + (size_t)row * R;
  float s = 0.0f;
  for (int i = tid; i < R; i += 256) { float v = xr[i]; s += v * v; }
  red[tid] = s;
  __syncthreads();
  for (int off = 128; off > 0; off >>= 1) {
    if (tid < off) red[tid] += red[tid + off];
    __syncthreads();
  }
  const float rs = rsqrtf(red[0] / (float)R + 1e-6f);
  _Float16* yr = Y + (size_t)row * R;
  for (int i = tid; i < R; i += 256) yr[i] = (_Float16)(xr[i] * rs * W[i]);
}

// ---------------- RoPE helpers + per-head repack ----------------
__device__ __forceinline__ void rope_pair(float x1, float x2, int s, int i,
                                          float* o1, float* o2) {
  const float ang = (float)s * __powf(10000.0f, -(float)(2 * i) / 64.0f);
  float sn, cs;
  __sincosf(ang, &sn, &cs);
  *o1 = x1 * cs - x2 * sn;
  *o2 = x1 * sn + x2 * cs;
}

// qn: [rows][NH*128], qp: [rows][NH*64]  ->  Qh: [b*NH+h][s][192]
__global__ void pack_q_kernel(const _Float16* __restrict__ qn,
                              const _Float16* __restrict__ qp,
                              _Float16* __restrict__ Qh) {
  const int idx = blockIdx.x * blockDim.x + threadIdx.x;
  const int total = kRows * kNH * (kDHead / 2);
  if (idx >= total) return;
  const int p = idx % (kDHead / 2);
  const int t = idx / (kDHead / 2);
  const int h = t % kNH;
  const int row = t / kNH;
  const int b = row / kS, s = row % kS;
  const size_t qbase = ((size_t)(b * kNH + h) * kS + s) * kDHead;
  const int d = 2 * p;
  if (d < kDNope) {
    Qh[qbase + d]     = qn[(size_t)row * (kNH * kDNope) + h * kDNope + d];
    Qh[qbase + d + 1] = qn[(size_t)row * (kNH * kDNope) + h * kDNope + d + 1];
  } else {
    const int dr = d - kDNope;
    float o1, o2;
    rope_pair((float)qp[(size_t)row * (kNH * kDRope) + h * kDRope + dr],
              (float)qp[(size_t)row * (kNH * kDRope) + h * kDRope + dr + 1],
              s, dr >> 1, &o1, &o2);
    Qh[qbase + d]     = (_Float16)o1;
    Qh[qbase + d + 1] = (_Float16)o2;
  }
}

// kvup: [rows][NH*256], krope: [rows][64] -> Kh: [b*NH+h][s][192]
__global__ void pack_k_kernel(const _Float16* __restrict__ kvup,
                              const _Float16* __restrict__ krope,
                              _Float16* __restrict__ Kh) {
  const int idx = blockIdx.x * blockDim.x + threadIdx.x;
  const int total = kRows * kNH * (kDHead / 2);
  if (idx >= total) return;
  const int p = idx % (kDHead / 2);
  const int t = idx / (kDHead / 2);
  const int h = t % kNH;
  const int row = t / kNH;
  const int b = row / kS, s = row % kS;
  const size_t kb = ((size_t)(b * kNH + h) * kS + s) * kDHead;
  const int d = 2 * p;
  if (d < kDNope) {
    Kh[kb + d]     = kvup[(size_t)row * (kNH * 256) + h * 256 + d];
    Kh[kb + d + 1] = kvup[(size_t)row * (kNH * 256) + h * 256 + d + 1];
  } else {
    const int dr = d - kDNope;
    float o1, o2;
    rope_pair((float)krope[(size_t)row * kDRope + dr],
              (float)krope[(size_t)row * kDRope + dr + 1], s, dr >> 1, &o1, &o2);
    Kh[kb + d]     = (_Float16)o1;
    Kh[kb + d + 1] = (_Float16)o2;
  }
}

// kvup -> VT: [b*NH+h][d][s] (transposed so P@V B-columns are contiguous)
__global__ void pack_v_kernel(const _Float16* __restrict__ kvup,
                              _Float16* __restrict__ VT) {
  const int idx = blockIdx.x * blockDim.x + threadIdx.x;
  const int total = kRows * kNH * kDV;
  if (idx >= total) return;
  const int d = idx % kDV;
  const int t = idx / kDV;
  const int h = t % kNH;
  const int row = t / kNH;
  const int b = row / kS, s = row % kS;
  VT[((size_t)(b * kNH + h) * kDV + d) * kS + s] =
      kvup[(size_t)row * (kNH * 256) + h * 256 + kDNope + d];
}

// ---------------- wave-level causal flash attention ----------------
// 4 independent waves per block; one wave owns 16 query rows of one head.
__global__ __launch_bounds__(128)
void mla_flash_attn_kernel(const _Float16* __restrict__ Qh,
                           const _Float16* __restrict__ Kh,
                           const _Float16* __restrict__ VT,
                           _Float16* __restrict__ O, float scale) {
  __shared__ _Float16 Pbuf[4][16][32];
  const int lane = threadIdx.x & 31;
  const int wv   = threadIdx.x >> 5;
  const int tiles_per_seq = kS / 16;                    // 128
  const int tile = blockIdx.x * 4 + wv;
  const int bh = tile / tiles_per_seq;
  const int q0 = (tile % tiles_per_seq) * 16;
  const int b = bh / kNH, h = bh % kNH;

  const _Float16* Q  = Qh + (size_t)bh * kS * kDHead;
  const _Float16* Kp = Kh + (size_t)bh * kS * kDHead;
  const _Float16* Vp = VT + (size_t)bh * kDV * kS;

  const int ml = lane & 15;
  const int hi = lane >> 4;

  // Q fragments (6 chunks of K=32) held in registers for the whole sweep
  v16h qf[6];
  {
    const _Float16* qrow = Q + (size_t)(q0 + ml) * kDHead;
    const int kh8 = hi * 8;
#pragma unroll
    for (int c = 0; c < 6; ++c)
      qf[c] = make_frag(*(const v8h*)(qrow + c * 32 + kh8),
                        *(const v8h*)(qrow + c * 32 + 16 + kh8));
  }

  v8f oacc[8];
#pragma unroll
  for (int dc = 0; dc < 8; ++dc) oacc[dc] = zero8f();
  float mrow[8], lrow[8];
#pragma unroll
  for (int s = 0; s < 8; ++s) { mrow[s] = -3.0e38f; lrow[s] = 0.0f; }

  const int nkb = (q0 + 16 + 31) / 32;   // 32-key blocks (causal bound)
  for (int kb = 0; kb < nkb; ++kb) {
    const int kbase = kb * 32;
    // ---- scores: 16 q x 32 keys, 6 WMMAs per 16-key half ----
    v8f sc[2];
#pragma unroll
    for (int h2 = 0; h2 < 2; ++h2) {
      sc[h2] = zero8f();
      const _Float16* krow = Kp + (size_t)(kbase + h2 * 16 + ml) * kDHead;
      const int ks16 = hi * 16;
#pragma unroll
      for (int c = 0; c < 6; ++c) {
        v16h bf = make_frag(*(const v8h*)(krow + c * 32 + hi * 8),
                            *(const v8h*)(krow + c * 32 + 16 + hi * 8));
        sc[h2] = __builtin_amdgcn_wmma_f32_16x16x32_f16(
            false, qf[c], false, bf, (short)0, sc[h2], false, false);
      }
    }
    // ---- scale + causal mask ----
#pragma unroll
    for (int h2 = 0; h2 < 2; ++h2) {
      const int key = kbase + h2 * 16 + ml;
#pragma unroll
      for (int s = 0; s < 8; ++s) {
        const int row = q0 + s + 8 * hi;
        const float v = sc[h2][s] * scale;
        sc[h2][s] = (key <= row) ? v : -1.0e30f;
      }
    }
    // ---- online softmax (row spans 16 lanes of a half-wave) ----
#pragma unroll
    for (int s = 0; s < 8; ++s) {
      float rm = fmaxf(sc[0][s], sc[1][s]);
#pragma unroll
      for (int off = 1; off < 16; off <<= 1)
        rm = fmaxf(rm, __shfl_xor(rm, off, 32));
      const float mnew = fmaxf(mrow[s], rm);
      const float alpha = __expf(mrow[s] - mnew);
      mrow[s] = mnew;
      const float p0 = __expf(sc[0][s] - mnew);
      const float p1 = __expf(sc[1][s] - mnew);
      sc[0][s] = p0; sc[1][s] = p1;
      float psum = p0 + p1;
#pragma unroll
      for (int off = 1; off < 16; off <<= 1)
        psum += __shfl_xor(psum, off, 32);
      lrow[s] = lrow[s] * alpha + psum;
#pragma unroll
      for (int dc = 0; dc < 8; ++dc) oacc[dc][s] *= alpha;
    }
    // ---- transpose P (C-layout -> A-layout) via wave-private LDS ----
#pragma unroll
    for (int h2 = 0; h2 < 2; ++h2)
#pragma unroll
      for (int s = 0; s < 8; ++s)
        Pbuf[wv][s + 8 * hi][h2 * 16 + ml] = (_Float16)sc[h2][s];
    v16h pf;
    {
      const _Float16* pr = &Pbuf[wv][ml][0];
      pf = make_frag(*(const v8h*)(pr + hi * 8),
                     *(const v8h*)(pr + 16 + hi * 8));
    }
    // ---- P @ V: 8 WMMAs, B-columns contiguous in VT rows ----
#pragma unroll
    for (int dc = 0; dc < 8; ++dc) {
      const _Float16* vrow = Vp + (size_t)(dc * 16 + ml) * kS + kbase + hi * 8;
      v16h bf = make_frag(*(const v8h*)(vrow), *(const v8h*)(vrow + 16));
      oacc[dc] = __builtin_amdgcn_wmma_f32_16x16x32_f16(
          false, pf, false, bf, (short)0, oacc[dc], false, false);
    }
  }
  // ---- normalize + write O[b*S+row][h*128 + d] ----
#pragma unroll
  for (int s = 0; s < 8; ++s) {
    const float inv = 1.0f / lrow[s];
    const int row = q0 + s + 8 * hi;
    _Float16* orow = O + ((size_t)(b * kS + row)) * (kNH * kDV) + h * kDV;
#pragma unroll
    for (int dc = 0; dc < 8; ++dc)
      orow[dc * 16 + ml] = (_Float16)(oacc[dc][s] * inv);
  }
}

// ---------------- host-side launch ----------------
static inline char* bump(char*& p, size_t bytes) {
  char* r = p;
  p += (bytes + 255) & ~(size_t)255;
  return r;
}

static inline void launch_gemm_f32(const _Float16* A, const _Float16* B, float* C,
                                   int M, int N, int K, hipStream_t st) {
  dim3 g((N + 127) / 128, (M + 127) / 128);
  gemm_nt_wmma_kernel<false><<<g, 256, 0, st>>>(A, B, (void*)C, M, N, K);
}
static inline void launch_gemm_f16(const _Float16* A, const _Float16* B,
                                   _Float16* C, int M, int N, int K,
                                   hipStream_t st) {
  dim3 g((N + 127) / 128, (M + 127) / 128);
  gemm_nt_wmma_kernel<true><<<g, 256, 0, st>>>(A, B, (void*)C, M, N, K);
}
static inline void launch_cvt(const float* s, _Float16* d, size_t n,
                              hipStream_t st) {
  f32_to_f16_kernel<<<2048, 256, 0, st>>>(s, d, n);
}

extern "C" void kernel_launch(void* const* d_in, const int* in_sizes, int n_in,
                              void* d_out, int out_size, void* d_ws,
                              size_t ws_size, hipStream_t stream) {
  const float* x        = (const float*)d_in[0];  // [B,S,DIM]
  const float* wq_down  = (const float*)d_in[1];  // [QRANK,DIM]
  const float* wq_up    = (const float*)d_in[2];  // [NH*DNOPE,QRANK]
  const float* wq_rope  = (const float*)d_in[3];  // [NH*DROPE,QRANK]
  const float* q_norm_w = (const float*)d_in[4];  // [QRANK]
  const float* wkv_down = (const float*)d_in[5];  // [KVRANK,DIM]
  const float* kv_norm_w= (const float*)d_in[6];  // [KVRANK]
  const float* wkv_up   = (const float*)d_in[7];  // [NH*256,KVRANK]
  const float* wk_rope  = (const float*)d_in[8];  // [DROPE,DIM]
  const float* wo       = (const float*)d_in[9];  // [DIM,NH*DV]
  float* out = (float*)d_out;

  char* ws = (char*)d_ws;
  _Float16* x16        = (_Float16*)bump(ws, (size_t)kRows * kDim * 2);
  _Float16* wq_down16  = (_Float16*)bump(ws, (size_t)kQRank * kDim * 2);
  _Float16* wq_up16    = (_Float16*)bump(ws, (size_t)kNH * kDNope * kQRank * 2);
  _Float16* wq_rope16  = (_Float16*)bump(ws, (size_t)kNH * kDRope * kQRank * 2);
  _Float16* wkv_down16 = (_Float16*)bump(ws, (size_t)kKVRank * kDim * 2);
  _Float16* wkv_up16   = (_Float16*)bump(ws, (size_t)kNH * 256 * kKVRank * 2);
  _Float16* wk_rope16  = (_Float16*)bump(ws, (size_t)kDRope * kDim * 2);
  _Float16* wo16       = (_Float16*)bump(ws, (size_t)kDim * kNH * kDV * 2);
  float*    qc_f32     = (float*)   bump(ws, (size_t)kRows * kQRank * 4);
  float*    kvc_f32    = (float*)   bump(ws, (size_t)kRows * kKVRank * 4);
  _Float16* qc16       = (_Float16*)bump(ws, (size_t)kRows * kQRank * 2);
  _Float16* kvc16      = (_Float16*)bump(ws, (size_t)kRows * kKVRank * 2);
  _Float16* qnope16    = (_Float16*)bump(ws, (size_t)kRows * kNH * kDNope * 2);
  _Float16* qpe16      = (_Float16*)bump(ws, (size_t)kRows * kNH * kDRope * 2);
  _Float16* kvup16     = (_Float16*)bump(ws, (size_t)kRows * kNH * 256 * 2);
  _Float16* krope16    = (_Float16*)bump(ws, (size_t)kRows * kDRope * 2);
  _Float16* Qh         = (_Float16*)bump(ws, (size_t)kB * kNH * kS * kDHead * 2);
  _Float16* Kh         = (_Float16*)bump(ws, (size_t)kB * kNH * kS * kDHead * 2);
  _Float16* VT         = (_Float16*)bump(ws, (size_t)kB * kNH * kDV * kS * 2);
  _Float16* O16        = (_Float16*)bump(ws, (size_t)kRows * kNH * kDV * 2);

  // 1) fp32 -> fp16
  launch_cvt(x,        x16,        (size_t)kRows * kDim, stream);
  launch_cvt(wq_down,  wq_down16,  (size_t)kQRank * kDim, stream);
  launch_cvt(wq_up,    wq_up16,    (size_t)kNH * kDNope * kQRank, stream);
  launch_cvt(wq_rope,  wq_rope16,  (size_t)kNH * kDRope * kQRank, stream);
  launch_cvt(wkv_down, wkv_down16, (size_t)kKVRank * kDim, stream);
  launch_cvt(wkv_up,   wkv_up16,   (size_t)kNH * 256 * kKVRank, stream);
  launch_cvt(wk_rope,  wk_rope16,  (size_t)kDRope * kDim, stream);
  launch_cvt(wo,       wo16,       (size_t)kDim * kNH * kDV, stream);

  // 2) down projections (fp32 out for norm precision) + shared rope-k
  launch_gemm_f32(x16, wq_down16,  qc_f32,  kRows, kQRank,  kDim, stream);
  launch_gemm_f32(x16, wkv_down16, kvc_f32, kRows, kKVRank, kDim, stream);
  launch_gemm_f16(x16, wk_rope16,  krope16, kRows, kDRope,  kDim, stream);

  // 3) RMS norms -> fp16
  rmsnorm_kernel<<<kRows, 256, 0, stream>>>(qc_f32,  q_norm_w,  qc16,  kQRank);
  rmsnorm_kernel<<<kRows, 256, 0, stream>>>(kvc_f32, kv_norm_w, kvc16, kKVRank);

  // 4) up projections
  launch_gemm_f16(qc16,  wq_up16,   qnope16, kRows, kNH * kDNope, kQRank, stream);
  launch_gemm_f16(qc16,  wq_rope16, qpe16,   kRows, kNH * kDRope, kQRank, stream);
  launch_gemm_f16(kvc16, wkv_up16,  kvup16,  kRows, kNH * 256,    kKVRank, stream);

  // 5) RoPE + per-head repack (V transposed)
  {
    const int tq = kRows * kNH * (kDHead / 2);
    pack_q_kernel<<<(tq + 255) / 256, 256, 0, stream>>>(qnope16, qpe16, Qh);
    pack_k_kernel<<<(tq + 255) / 256, 256, 0, stream>>>(kvup16, krope16, Kh);
    const int tv = kRows * kNH * kDV;
    pack_v_kernel<<<(tv + 255) / 256, 256, 0, stream>>>(kvup16, VT);
  }

  // 6) causal flash attention (4 waves / block, 16-query tiles)
  {
    const int tiles = kB * kNH * (kS / 16);           // 4096
    const float scale = 1.0f / sqrtf((float)kDHead);  // 1/sqrt(192)
    mla_flash_attn_kernel<<<tiles / 4, 128, 0, stream>>>(Qh, Kh, VT, O16, scale);
  }

  // 7) output projection -> fp32 result
  launch_gemm_f32(O16, wo16, out, kRows, kDim, kNH * kDV, stream);
  (void)in_sizes; (void)n_in; (void)out_size; (void)ws_size;
}
